// ClusteringLayer_60052232732701
// MI455X (gfx1250) — matrix operand order repaired and checked by reading, not compiled, not run on hardware
//
#include <hip/hip_runtime.h>
#include <hip/hip_bf16.h>

typedef __attribute__((ext_vector_type(16))) _Float16 v16h;
typedef __attribute__((ext_vector_type(8)))  _Float16 v8h;
typedef __attribute__((ext_vector_type(4)))  _Float16 v4h;
typedef __attribute__((ext_vector_type(8)))  float    v8f;

#define DDIM 1024
#define KOUT 256
#define TM   16     // rows per block
#define KC   128    // k-chunk staged in LDS
#define XA_STRIDE (KC + 8)   // f16 elements, padded against bank conflicts
#define CB_STRIDE (KC + 8)
#define EPSN 1e-8f

// ---------------- Kernel 1: normalize centers -> f16 in workspace ----------
__global__ __launch_bounds__(256)
void cl_normalize_centers(const float* __restrict__ cents,
                          _Float16* __restrict__ cn) {
    __shared__ float wsum[8];
    const int row = blockIdx.x;          // one center per block
    const int t   = threadIdx.x;         // 256 threads, 4 floats each
    const float4 v = reinterpret_cast<const float4*>(cents + (size_t)row * DDIM)[t];
    float s = v.x * v.x + v.y * v.y + v.z * v.z + v.w * v.w;
    #pragma unroll
    for (int off = 16; off > 0; off >>= 1) s += __shfl_xor(s, off, 32);
    if ((t & 31) == 0) wsum[t >> 5] = s;
    __syncthreads();
    float total = 0.f;
    #pragma unroll
    for (int i = 0; i < 8; ++i) total += wsum[i];
    const float rn = 1.0f / fmaxf(sqrtf(total), EPSN);
    v4h h;
    h[0] = (_Float16)(v.x * rn);
    h[1] = (_Float16)(v.y * rn);
    h[2] = (_Float16)(v.z * rn);
    h[3] = (_Float16)(v.w * rn);
    reinterpret_cast<v4h*>(cn + (size_t)row * DDIM)[t] = h;   // 8B store
}

// ---------------- Kernel 2: normalized GEMM via WMMA -----------------------
// block = 256 threads (8 waves). Block tile: 16 rows x 256 cols.
// x staged RAW as f16; per-thread sumsq accumulated during staging; the
// 1/||x|| factor is applied per output row in the epilogue (c is already
// unit-norm), so x is read from global exactly once.
__global__ __launch_bounds__(256)
void cl_cosine_wmma(const float* __restrict__ x,
                    const _Float16* __restrict__ cn,
                    float* __restrict__ out) {
    __shared__ _Float16 sXA[TM * XA_STRIDE];     // 4,352 B
    __shared__ _Float16 sCB[KOUT * CB_STRIDE];   // 69,632 B
    __shared__ float    sRnorm[TM];

    const int t    = threadIdx.x;
    const int r    = t >> 4;        // 0..15 : row within tile
    const int l16  = t & 15;        // 0..15 : sub-lane within row group
    const size_t grow = (size_t)blockIdx.x * TM + r;
    const float* xr = x + grow * DDIM;

    const int wave = t >> 5;
    const int lane = t & 31;
    const int half = lane >> 4;     // 0 | 1 : which half-wave (K sub-range)
    const int lm   = lane & 15;     // M for A-frag, N for B-frag

    v8f acc0 = {};
    v8f acc1 = {};
    float ssq = 0.f;                // partial sum of x^2 for row r

    for (int kc = 0; kc < DDIM; kc += KC) {
        // ---- stage raw x chunk: 16 x 128 f32 -> f16, 2 float4/thread;
        //      fold the sumsq accumulation into the same pass.
        #pragma unroll
        for (int j = 0; j < 2; ++j) {
            const int c4 = l16 + j * 16;               // float4 index in row
            const float4 w = reinterpret_cast<const float4*>(xr + kc)[c4];
            ssq += w.x * w.x + w.y * w.y + w.z * w.z + w.w * w.w;
            v4h h;
            h[0] = (_Float16)w.x;
            h[1] = (_Float16)w.y;
            h[2] = (_Float16)w.z;
            h[3] = (_Float16)w.w;
            *reinterpret_cast<v4h*>(sXA + r * XA_STRIDE + c4 * 4) = h;
        }
        // ---- stage centers chunk: 256 x 128 f16, 8 f16 per thread per iter
        #pragma unroll
        for (int it = 0; it < 16; ++it) {
            const int idx = it * 2048 + t * 8;
            const int n   = idx >> 7;       // center row
            const int c   = idx & 127;      // k within chunk
            const v8h hv = *reinterpret_cast<const v8h*>(cn + (size_t)n * DDIM + kc + c);
            *reinterpret_cast<v8h*>(sCB + n * CB_STRIDE + c) = hv;
        }
        // ---- prefetch next x chunk (covers every cacheline of the 16x128 tile)
        if (kc + KC < DDIM) {
            __builtin_prefetch(xr + kc + KC + l16 * 8, 0, 3);
        }
        __syncthreads();

        // ---- 4 WMMA k-steps of 32 over this chunk
        #pragma unroll
        for (int kk = 0; kk < 4; ++kk) {
            // A fragment (16x32 f16): lane lm = M; K = kk*32 + half*16 + 0..15
            const _Float16* ap = sXA + lm * XA_STRIDE + kk * 32 + half * 16;
            const v8h alo = *reinterpret_cast<const v8h*>(ap);
            const v8h ahi = *reinterpret_cast<const v8h*>(ap + 8);
            v16h a;
            #pragma unroll
            for (int i = 0; i < 8; ++i) { a[i] = alo[i]; a[i + 8] = ahi[i]; }

            // B fragments (32x16 f16): lane lm = N; K = kk*32 + half*16 + 0..15
            {
                const _Float16* bp = sCB + ((wave * 2 + 0) * 16 + lm) * CB_STRIDE
                                         + kk * 32 + half * 16;
                const v8h blo = *reinterpret_cast<const v8h*>(bp);
                const v8h bhi = *reinterpret_cast<const v8h*>(bp + 8);
                v16h b;
                #pragma unroll
                for (int i = 0; i < 8; ++i) { b[i] = blo[i]; b[i + 8] = bhi[i]; }
                acc0 = __builtin_amdgcn_wmma_f32_16x16x32_f16(
                    false, a, false, b, (short)0, acc0, false, false);
            }
            {
                const _Float16* bp = sCB + ((wave * 2 + 1) * 16 + lm) * CB_STRIDE
                                         + kk * 32 + half * 16;
                const v8h blo = *reinterpret_cast<const v8h*>(bp);
                const v8h bhi = *reinterpret_cast<const v8h*>(bp + 8);
                v16h b;
                #pragma unroll
                for (int i = 0; i < 8; ++i) { b[i] = blo[i]; b[i + 8] = bhi[i]; }
                acc1 = __builtin_amdgcn_wmma_f32_16x16x32_f16(
                    false, a, false, b, (short)0, acc1, false, false);
            }
        }
        __syncthreads();
    }

    // ---- finish row norms: reduce the 16 partials of each row
    #pragma unroll
    for (int off = 8; off > 0; off >>= 1) ssq += __shfl_xor(ssq, off, 16);
    if (l16 == 0) sRnorm[r] = 1.0f / fmaxf(sqrtf(ssq), EPSN);
    __syncthreads();

    // ---- epilogue: (dot * rnorm[M] + 1) * 0.5, store f32.
    //      C/D layout: VGPR rr -> M = rr + 8*half, N = lane%16.
    const size_t rowBase = (size_t)blockIdx.x * TM;
    #pragma unroll
    for (int rr = 0; rr < 8; ++rr) {
        const int M = rr + 8 * half;
        const float rnM = sRnorm[M];
        const size_t o = (rowBase + M) * KOUT + lm;
        out[o + (wave * 2 + 0) * 16] = (acc0[rr] * rnM + 1.0f) * 0.5f;
        out[o + (wave * 2 + 1) * 16] = (acc1[rr] * rnM + 1.0f) * 0.5f;
    }
}

extern "C" void kernel_launch(void* const* d_in, const int* in_sizes, int n_in,
                              void* d_out, int out_size, void* d_ws, size_t ws_size,
                              hipStream_t stream) {
    const float* x     = (const float*)d_in[0];     // [64,512,1024] f32
    const float* cents = (const float*)d_in[1];     // [256,1024]   f32
    float* out = (float*)d_out;                     // [64,512,256] f32
    _Float16* cn = (_Float16*)d_ws;                 // 256*1024*2 = 512 KB scratch

    const int N = in_sizes[0] / DDIM;               // 32768 rows

    cl_normalize_centers<<<KOUT, 256, 0, stream>>>(cents, cn);
    cl_cosine_wmma<<<N / TM, 256, 0, stream>>>(x, cn, out);
}
